// EncoderLinearQuery_31250182046440
// MI455X (gfx1250) — matrix operand order repaired and checked by reading, not compiled, not run on hardware
//
#include <hip/hip_runtime.h>
#include <hip/hip_bf16.h>

// ---------------------------------------------------------------------------
// CDNA5 / gfx1250 implementation.
// Pipeline:
//   1) prep_weights : f32 -> bf16 copies of wih_d, wih_p, w_lin, whh_d, whh_p
//   2) gather_mean  : embedding gather+mean -> seq_d/seq_p (bf16)
//                     (reference quirk: BOTH gathers read emb_diag)
//   3) gi_gemm      : gi = seq @ wih^T + bih  (WMMA bf16, f32 accum)
//   4) gru_scan     : sequential GRU; r/z weights TDM-loaded into LDS (bf16),
//                     n-gate weights live in VGPR A-fragments; 24 v_wmma/step
//   5) final_gemm   : queries = relu(patient) @ w_lin^T + b_lin (WMMA)
//   6) meds_copy    : int medication codes -> float tail of d_out
// ---------------------------------------------------------------------------

typedef __attribute__((ext_vector_type(16))) __bf16    v16bf;
typedef __attribute__((ext_vector_type(8)))  float     v8f;
typedef __attribute__((ext_vector_type(4)))  unsigned  v4u;
typedef __attribute__((ext_vector_type(4)))  int       v4i;
typedef __attribute__((ext_vector_type(8)))  int       v8i;

#define T_LEN 4096
#define HDIM  256

__device__ inline v8f wmma_bf16(v16bf a, v16bf b, v8f c) {
  return __builtin_amdgcn_wmma_f32_16x16x32_bf16(
      /*neg_a=*/false, a, /*neg_b=*/false, b,
      /*c_mod=*/(short)0, c, /*reuse_a=*/false, /*reuse_b=*/false);
}

// A-matrix 16x32 bf16 fragment (ISA 7.12.2): lane L holds row M = L%16.
// lanes 0-15 carry K = {0..7, 16..23}; lanes 16-31 carry K = {8..15, 24..31}.
__device__ inline v16bf load_a_frag(const unsigned* base_dw, int row,
                                    int rowStrideDw, int kk, int laneHi) {
  union { v4u q[2]; v16bf v; } u;
  const v4u* p = (const v4u*)(base_dw + row * rowStrideDw + kk * 16 + (laneHi ? 4 : 0));
  u.q[0] = p[0];  // dwords +0..3   -> K pairs 0..7  (or 8..15)
  u.q[1] = p[2];  // dwords +8..11  -> K pairs 16..23 (or 24..31)
  return u.v;
}

// B-matrix 32x16 bf16 fragment: lane L holds column N = L%16.
// lanes 0-15 carry K = {0..7, 16..23}; lanes 16-31 carry K = {8..15, 24..31}.
// Source is row-major [N][K] (i.e. W with gh = W @ h -> B[k][n] = W[n][k]).
__device__ inline v16bf load_b_frag(const unsigned* base_dw, int n,
                                    int rowStrideDw, int kk, int laneHi) {
  union { v4u q[2]; v16bf v; } u;
  const v4u* p = (const v4u*)(base_dw + n * rowStrideDw + kk * 16 + (laneHi ? 4 : 0));
  u.q[0] = p[0];
  u.q[1] = p[2];
  return u.v;
}

// ---------------------------------------------------------------------------
// 1) f32 -> bf16 weight conversion (wih/whh for both GRUs + w_lin)
__global__ void prep_weights(const float* wih_d, const float* wih_p, const float* w_lin,
                             const float* whh_d, const float* whh_p,
                             __bf16* wih_d_bf, __bf16* wih_p_bf, __bf16* wlin_bf,
                             __bf16* whh_d_bf, __bf16* whh_p_bf) {
  int i = blockIdx.x * 256 + threadIdx.x;
  if (i < 196608) {
    wih_d_bf[i] = (__bf16)wih_d[i];
    wih_p_bf[i] = (__bf16)wih_p[i];
    whh_d_bf[i] = (__bf16)whh_d[i];
    whh_p_bf[i] = (__bf16)whh_p[i];
  }
  if (i < 131072) wlin_bf[i] = (__bf16)w_lin[i];
}

// ---------------------------------------------------------------------------
// 2) embedding gather + mean  (both from emb_diag, as in the reference)
__global__ void gather_mean(const float* __restrict__ emb,
                            const int* __restrict__ diag_codes,
                            const int* __restrict__ proc_codes,
                            __bf16* seq_d, __bf16* seq_p) {
  int t = blockIdx.x;
  int e = threadIdx.x;
  int which = blockIdx.y;           // 0 = diag(32 codes), 1 = proc(16 codes)
  const int* codes = which ? proc_codes : diag_codes;
  int cnt = which ? 16 : 32;
  const int* row = codes + t * cnt;
  float s = 0.0f;
  for (int i = 0; i < cnt; ++i) s += emb[(long)row[i] * HDIM + e];
  s *= which ? (1.0f / 16.0f) : (1.0f / 32.0f);
  (which ? seq_p : seq_d)[t * HDIM + e] = (__bf16)s;
}

// ---------------------------------------------------------------------------
// 3) gi = seq @ wih^T + bih, for both GRUs.  M=4096, N=768, K=256.
//    One wave per 16x16 tile; 8 waves/block; 24576 tiles total.
__global__ void gi_gemm(const unsigned* seq_d, const unsigned* seq_p,
                        const unsigned* wih_d_bf, const unsigned* wih_p_bf,
                        const float* bih_d, const float* bih_p,
                        float* gi_d, float* gi_p) {
  int tid  = threadIdx.x;
  int lane = tid & 31, wave = tid >> 5;
  int tile = blockIdx.x * 8 + wave;            // [0, 24576)
  int z    = tile / (256 * 48);
  int rem  = tile % (256 * 48);
  int mt   = rem / 48, nt = rem % 48;
  const unsigned* A    = z ? seq_p    : seq_d;
  const unsigned* B    = z ? wih_p_bf : wih_d_bf;
  const float*    bias = z ? bih_p    : bih_d;
  float*          out  = z ? gi_p     : gi_d;
  int laneHi = lane >> 4, l = lane & 15;
  int mrow = mt * 16 + l;
  int nrow = nt * 16 + l;
  v8f c = {};
#pragma unroll
  for (int kk = 0; kk < 8; ++kk) {             // K = 256 = 8 * 32
    v16bf a = load_a_frag(A, mrow, 128, kk, laneHi);
    v16bf b = load_b_frag(B, nrow, 128, kk, laneHi);
    c = wmma_bf16(a, b, c);
  }
  float bv = bias[nt * 16 + l];
#pragma unroll
  for (int r = 0; r < 8; ++r) {
    int m = mt * 16 + r + laneHi * 8;
    out[(long)m * 768 + nt * 16 + l] = c[r] + bv;
  }
}

// ---------------------------------------------------------------------------
// 4) sequential GRU scan. grid.x = 2 (z=0: diag GRU, z=1: proc GRU),
//    512 threads = 16 waves. Dynamic LDS:
//      [0        , 262144) : whh rows 0..511 (r,z gates) as bf16 (TDM-loaded)
//      [262144   , 265216) : gh[768] f32
//      [265216   , 265728) : h[256] bf16
//    n-gate weights (rows 512..767) live in VGPRs as prebuilt A-fragments.
__global__ void __launch_bounds__(512)
gru_scan(const float* __restrict__ whh_d, const float* __restrict__ bhh_d,
         const float* __restrict__ gi_d,
         const float* __restrict__ whh_p, const float* __restrict__ bhh_p,
         const float* __restrict__ gi_p,
         const __bf16* __restrict__ whh_d_bf, const __bf16* __restrict__ whh_p_bf,
         __bf16* patient_relu) {
  extern __shared__ char smem[];
  __bf16* w_rz = (__bf16*)smem;                    // 512*256 bf16 = 256KB
  float*  gh   = (float*)(smem + 262144);          // 768 f32
  __bf16* h_bf = (__bf16*)(smem + 262144 + 3072);  // 256 bf16

  int zsel = blockIdx.x;
  const float*  whh    = zsel ? whh_p    : whh_d;
  const float*  bhh    = zsel ? bhh_p    : bhh_d;
  const float*  gi     = zsel ? gi_p     : gi_d;
  const __bf16* whh_bf = zsel ? whh_p_bf : whh_d_bf;

  int tid = threadIdx.x;
  int lane = tid & 31, wave = tid >> 5;
  int laneHi = lane >> 4, l = lane & 15;

  // --- TDM preload: r/z gate weight block (512x256 bf16 = 256KB, contiguous)
  //     global(bf16 image of whh rows 0..511) -> LDS, as one 2D descriptor:
  //     16 rows x 4096 dwords, stride 4096 dwords. Issued by wave 0 only
  //     (TDM ignores EXEC; one descriptor per workgroup).
  if (tid < 32) {
    unsigned lds_off = (unsigned)(unsigned long long)(void*)smem;  // low 32 bits = LDS byte offset
    unsigned long long ga = (unsigned long long)(void*)whh_bf;
    v4i g0;
    g0[0] = 1;                                        // count=1 (valid), no gather
    g0[1] = (int)lds_off;                             // lds_addr
    g0[2] = (int)(ga & 0xffffffffull);                // global_addr[31:0]
    g0[3] = (int)((unsigned)(ga >> 32) | (2u << 30)); // global_addr[56:32] | type=2
    v8i g1;
    g1[0] = 0x00020000;                // data_size=2 -> 4-byte elements
    g1[1] = (int)(4096u << 16);        // tensor_dim0[15:0] = 4096 dwords/row
    g1[2] = (int)(16u << 16);          // tensor_dim1[15:0] = 16 rows
    g1[3] = (int)(4096u << 16);        // tile_dim0 = 4096 dwords
    g1[4] = 16;                        // tile_dim1 = 16 rows
    g1[5] = 4096;                      // tensor_dim0_stride = 4096 dwords
    g1[6] = 0;
    g1[7] = 0;
    asm volatile("tensor_load_to_lds %0, %1" :: "s"(g0), "s"(g1) : "memory");
    __builtin_amdgcn_s_wait_tensorcnt(0);
  }

  // prebuild n-gate A-fragments (rows 512 + wave*16 + l) in registers
  v16bf an[8];
  {
    int row = 512 + wave * 16 + l;
#pragma unroll
    for (int kk = 0; kk < 8; ++kk) {
      union { __bf16 h[16]; v16bf v; } u;
#pragma unroll
      for (int j = 0; j < 8; ++j) {
        int k = kk * 32 + (laneHi ? 8 : 0) + (j < 4 ? 2 * j : 16 + 2 * (j - 4));
        u.h[2 * j]     = (__bf16)whh[(long)row * 256 + k];
        u.h[2 * j + 1] = (__bf16)whh[(long)row * 256 + k + 1];
      }
      an[kk] = u.v;
    }
  }

  if (tid < 256) h_bf[tid] = (__bf16)0.0f;
  float h = 0.0f;
  __syncthreads();

  const unsigned* wrz_dw = (const unsigned*)w_rz;
  const unsigned* h_dw   = (const unsigned*)h_bf;

  for (int t = 0; t < T_LEN; ++t) {
    v8f c0 = {}, c1 = {}, c2 = {};
#pragma unroll
    for (int kk = 0; kk < 8; ++kk) {
      // B fragment: column 0 = current h (bf16), other columns zero.
      union { v4u q[2]; v16bf v; } ub;
      ub.q[0] = (v4u)0; ub.q[1] = (v4u)0;
      if (l == 0) {
        const v4u* hp = (const v4u*)(h_dw + kk * 16 + laneHi * 4);
        ub.q[0] = hp[0];
        ub.q[1] = hp[2];
      }
      v16bf b  = ub.v;
      v16bf a0 = load_a_frag(wrz_dw, wave * 16 + l,        128, kk, laneHi); // r gate
      v16bf a1 = load_a_frag(wrz_dw, 256 + wave * 16 + l,  128, kk, laneHi); // z gate
      c0 = wmma_bf16(a0, b, c0);
      c1 = wmma_bf16(a1, b, c1);
      c2 = wmma_bf16(an[kk], b, c2);                                        // n gate
    }
    // D column 0 lives in lanes 0 (M=0..7) and 16 (M=8..15)
    if (l == 0) {
      int mo = laneHi * 8;
#pragma unroll
      for (int r = 0; r < 8; ++r) {
        gh[wave * 16 + mo + r]       = c0[r];
        gh[256 + wave * 16 + mo + r] = c1[r];
        gh[512 + wave * 16 + mo + r] = c2[r];
      }
    }
    __syncthreads();
    if (tid < 256) {
      int j = tid;
      const float* girow = gi + (long)t * 768;
      float ir = girow[j], iz = girow[256 + j], inn = girow[512 + j];
      float gr = gh[j]       + bhh[j];
      float gz = gh[256 + j] + bhh[256 + j];
      float gn = gh[512 + j] + bhh[512 + j];
      float r = 1.0f / (1.0f + __expf(-(ir + gr)));
      float z = 1.0f / (1.0f + __expf(-(iz + gz)));
      float a = inn + r * gn;
      float e2 = __expf(2.0f * a);             // tanh via exp (transcendental unit)
      float n = (e2 - 1.0f) / (e2 + 1.0f);
      h = (1.0f - z) * n + z * h;
      h_bf[j] = (__bf16)h;
      // store relu(h) in bf16: cols [0,256) = diag GRU, [256,512) = proc GRU
      patient_relu[(long)t * 512 + zsel * 256 + j] = (__bf16)fmaxf(h, 0.0f);
      if (t + 1 < T_LEN) __builtin_prefetch(girow + 768, 0, 0);
    }
    __syncthreads();
  }
}

// ---------------------------------------------------------------------------
// 5) queries = relu(patient) @ w_lin^T + b_lin.  M=4096, K=512, N=256.
//    Writes directly into d_out with query/memory_keys split.
__global__ void final_gemm(const unsigned* patient_bf, const unsigned* wlin_bf,
                           const float* b_lin, float* d_out) {
  int tid  = threadIdx.x;
  int lane = tid & 31, wave = tid >> 5;
  int tile = blockIdx.x * 8 + wave;           // [0, 4096)
  int mt = tile / 16, nt = tile % 16;
  int laneHi = lane >> 4, l = lane & 15;
  int mrow = mt * 16 + l;
  int nrow = nt * 16 + l;
  v8f c = {};
#pragma unroll
  for (int kk = 0; kk < 16; ++kk) {           // K = 512 = 16 * 32
    v16bf a = load_a_frag(patient_bf, mrow, 256, kk, laneHi);
    v16bf b = load_b_frag(wlin_bf,    nrow, 256, kk, laneHi);
    c = wmma_bf16(a, b, c);
  }
  float bv = b_lin[nt * 16 + l];
#pragma unroll
  for (int r = 0; r < 8; ++r) {
    int m = mt * 16 + r + laneHi * 8;
    int n = nt * 16 + l;
    float v = c[r] + bv;
    if (m == T_LEN - 1)
      d_out[n] = v;                                  // query (row 4095)
    else
      d_out[256 + (long)m * 256 + n] = v;            // memory_keys rows 0..4094
  }
}

// ---------------------------------------------------------------------------
// 6) memory_values = input_medication[:-1] as float
__global__ void meds_copy(const int* __restrict__ meds, float* d_out) {
  int i = blockIdx.x * 256 + threadIdx.x;
  if (i < 4095 * 24) d_out[1048576 + i] = (float)meds[i];
}

// ---------------------------------------------------------------------------
extern "C" void kernel_launch(void* const* d_in, const int* in_sizes, int n_in,
                              void* d_out, int out_size, void* d_ws, size_t ws_size,
                              hipStream_t stream) {
  const float* emb_diag = (const float*)d_in[0];
  // d_in[1] (emb_proc) intentionally unused: reference gathers both from emb_diag
  const float* wih_d = (const float*)d_in[2];
  const float* whh_d = (const float*)d_in[3];
  const float* bih_d = (const float*)d_in[4];
  const float* bhh_d = (const float*)d_in[5];
  const float* wih_p = (const float*)d_in[6];
  const float* whh_p = (const float*)d_in[7];
  const float* bih_p = (const float*)d_in[8];
  const float* bhh_p = (const float*)d_in[9];
  const float* w_lin = (const float*)d_in[10];
  const float* b_lin = (const float*)d_in[11];
  const int* diag_codes = (const int*)d_in[12];
  const int* proc_codes = (const int*)d_in[13];
  const int* meds       = (const int*)d_in[14];
  float* out = (float*)d_out;
  char*  ws  = (char*)d_ws;

  // workspace layout (bytes, all 256-aligned); total ~35.4 MB
  __bf16* seq_d     = (__bf16*)(ws + 0);
  __bf16* seq_p     = (__bf16*)(ws + 2097152);
  __bf16* wih_d_bf  = (__bf16*)(ws + 4194304);
  __bf16* wih_p_bf  = (__bf16*)(ws + 4587520);
  __bf16* wlin_bf   = (__bf16*)(ws + 4980736);
  float*  gi_d      = (float*) (ws + 5242880);
  float*  gi_p      = (float*) (ws + 17825792);
  __bf16* patient   = (__bf16*)(ws + 30408704);
  __bf16* whh_d_bf  = (__bf16*)(ws + 34603008);
  __bf16* whh_p_bf  = (__bf16*)(ws + 34996224);

  prep_weights<<<768, 256, 0, stream>>>(wih_d, wih_p, w_lin, whh_d, whh_p,
                                        wih_d_bf, wih_p_bf, wlin_bf,
                                        whh_d_bf, whh_p_bf);

  gather_mean<<<dim3(T_LEN, 2), 256, 0, stream>>>(emb_diag, diag_codes, proc_codes,
                                                  seq_d, seq_p);

  gi_gemm<<<3072, 256, 0, stream>>>((const unsigned*)seq_d, (const unsigned*)seq_p,
                                    (const unsigned*)wih_d_bf, (const unsigned*)wih_p_bf,
                                    bih_d, bih_p, gi_d, gi_p);

  const int scan_lds = 262144 + 3072 + 512;   // 265728 bytes (< 320KB CDNA5 limit)
  (void)hipFuncSetAttribute(reinterpret_cast<const void*>(gru_scan),
                            hipFuncAttributeMaxDynamicSharedMemorySize, scan_lds);
  gru_scan<<<2, 512, scan_lds, stream>>>(whh_d, bhh_d, gi_d,
                                         whh_p, bhh_p, gi_p,
                                         whh_d_bf, whh_p_bf, patient);

  final_gemm<<<512, 256, 0, stream>>>((const unsigned*)patient,
                                      (const unsigned*)wlin_bf, b_lin, out);

  meds_copy<<<(4095 * 24 + 255) / 256, 256, 0, stream>>>(meds, out);
}
